// RNNDecoder_2035814498665
// MI455X (gfx1250) — matrix operand order repaired and checked by reading, not compiled, not run on hardware
//
#include <hip/hip_runtime.h>

#define TT 128
#define BB 16
#define EE 512
#define HH 512
#define VV 32000
#define G4H 2048   // 4*H

typedef __attribute__((ext_vector_type(16))) __bf16 v16bf;
typedef __attribute__((ext_vector_type(8)))  float  v8f;

// sched_group_barrier masks
#define SG_VMEM_READ 0x020
#define SG_WMMA      0x008

union FragBF {
    v16bf v;
    uint4 q[2];
};

__device__ __forceinline__ unsigned short f32_to_bf16(float f) {
    unsigned int u = __float_as_uint(f);
    unsigned int lsb = (u >> 16) & 1u;
    u += 0x7fffu + lsb;              // round-to-nearest-even
    return (unsigned short)(u >> 16);
}

// A-matrix fragment: 16x32 bf16 tile of row-major matrix A[*, lda].
// lane&15 = M row; lanes 0-15: K = k0+{0..7,16..23}; lanes 16-31: +8.
__device__ __forceinline__ v16bf load_a_frag(const unsigned short* __restrict__ A,
                                             int lda, int m_base, int k0, int lane) {
    int m    = m_base + (lane & 15);
    int koff = (lane >> 4) << 3;     // 0 or 8
    const unsigned short* p = A + (size_t)m * lda + k0 + koff;
    FragBF f;
    f.q[0] = *(const uint4*)(p);         // K chunk 0: 8 bf16
    f.q[1] = *(const uint4*)(p + 16);    // K chunk 1: 8 bf16
    return f.v;
}

// B-matrix fragment: 32x16 tile where B[k][n] = W[n_global][k], W row-major [*, ldw].
// lane&15 = N col; lanes 0-15: K = k0+{0..7,16..23}; lanes 16-31: +8.
__device__ __forceinline__ v16bf load_b_frag(const unsigned short* __restrict__ W,
                                             int ldw, int n_base, int k0, int lane) {
    int n    = n_base + (lane & 15);
    int koff = (lane >> 4) << 3;     // 0 or 8
    const unsigned short* p = W + (size_t)n * ldw + k0 + koff;
    FragBF f;
    f.q[0] = *(const uint4*)(p);
    f.q[1] = *(const uint4*)(p + 16);
    return f.v;
}

__device__ __forceinline__ float sigmoidf_(float x) { return 1.0f / (1.0f + __expf(-x)); }

// ---------------- f32 -> bf16 bulk convert ----------------
__global__ void cvt_bf16_kernel(const float* __restrict__ src,
                                unsigned short* __restrict__ dst, int n) {
    int i = blockIdx.x * blockDim.x + threadIdx.x;
    int stride = gridDim.x * blockDim.x;
    for (; i < n; i += stride) dst[i] = f32_to_bf16(src[i]);
}

// ---------------- embedding gather -> time-major bf16 [T,B,E] ----------------
__global__ void embed_kernel(const int* __restrict__ xs,       // [B,T]
                             const float* __restrict__ emb,    // [V,E]
                             unsigned short* __restrict__ xemb) {
    int idx = blockIdx.x * blockDim.x + threadIdx.x;           // T*B*E threads
    int e  = idx & (EE - 1);
    int tb = idx >> 9;                                         // log2(E)=9
    int b  = tb & (BB - 1);
    int t  = tb >> 4;                                          // log2(B)=4
    if (t >= TT) return;
    int tok = xs[b * TT + t];
    xemb[idx] = f32_to_bf16(emb[(size_t)tok * EE + e]);
}

// ---------------- state init ----------------
__global__ void init_state_kernel(const float* __restrict__ enc_h,
                                  const float* __restrict__ enc_c,
                                  unsigned short* __restrict__ h0,
                                  float* __restrict__ c, int n) {
    int i = blockIdx.x * blockDim.x + threadIdx.x;
    if (i >= n) return;
    h0[i] = f32_to_bf16(enc_h[i]);
    c[i]  = enc_c[i];
}

// ---------------- one LSTM timestep ----------------
// grid = H/16 = 32 blocks x 128 threads (4 waves). Block owns hidden cols
// [16*blk, 16*blk+16). K = E + H = 1024 split into four 256-wide quarters,
// one per wave. 8 K-steps fully unrolled, double register buffers, and
// sched_group_barrier pins the schedule to [10 loads (it+1)] -> [4 WMMA (it)]
// so each WMMA group waits behind a partial loadcnt, not zero.
__global__ __launch_bounds__(128) void lstm_step_kernel(
    const unsigned short* __restrict__ xt,     // [B,E] bf16 (xemb + t*B*E)
    const unsigned short* __restrict__ Wih,    // [4H,E] bf16
    const unsigned short* __restrict__ Whh,    // [4H,H] bf16
    const float* __restrict__ b_ih,            // [4H]
    const float* __restrict__ b_hh,            // [4H]
    const unsigned short* __restrict__ h_in,   // [B,H] bf16
    unsigned short* __restrict__ h_out,        // [B,H] bf16
    float* __restrict__ c,                     // [B,H] f32, in/out (block-owned cols)
    unsigned short* __restrict__ hs_out)       // cat_hs bf16 + t*B*H
{
    __shared__ float red[4][4][256];           // [wave][gate][reg*32+lane] = 16 KB

    const int lane = threadIdx.x & 31;
    const int wave = threadIdx.x >> 5;         // 0..3 = K quarter
    const int col0 = blockIdx.x * 16;

    const unsigned short* Aptr = (wave < 2) ? xt  : h_in;
    const unsigned short* Wptr = (wave < 2) ? Wih : Whh;
    const int kq = (wave & 1) * 256;           // quarter base within its matrix

    v8f acc[4] = {v8f{}, v8f{}, v8f{}, v8f{}};

    v16bf a_buf[2];
    v16bf b_buf[2][4];

    a_buf[0] = load_a_frag(Aptr, 512, 0, kq, lane);
#pragma unroll
    for (int g = 0; g < 4; ++g)
        b_buf[0][g] = load_b_frag(Wptr, 512, g * HH + col0, kq, lane);
    __builtin_amdgcn_sched_group_barrier(SG_VMEM_READ, 10, 0);   // prologue loads

#pragma unroll
    for (int it = 0; it < 8; ++it) {
        const int cur = it & 1;
        const int nxt = cur ^ 1;
        if (it < 7) {
            const int k0 = kq + (it + 1) * 32;
            a_buf[nxt] = load_a_frag(Aptr, 512, 0, k0, lane);
#pragma unroll
            for (int g = 0; g < 4; ++g)
                b_buf[nxt][g] = load_b_frag(Wptr, 512, g * HH + col0, k0, lane);
        }
#pragma unroll
        for (int g = 0; g < 4; ++g)
            acc[g] = __builtin_amdgcn_wmma_f32_16x16x32_bf16(
                false, a_buf[cur], false, b_buf[cur][g], (short)0, acc[g], false, false);
        if (it < 7) __builtin_amdgcn_sched_group_barrier(SG_VMEM_READ, 10, 0);
        __builtin_amdgcn_sched_group_barrier(SG_WMMA, 4, 0);
    }

    // stash partials in LDS
#pragma unroll
    for (int g = 0; g < 4; ++g)
#pragma unroll
        for (int r = 0; r < 8; ++r)
            red[wave][g][r * 32 + lane] = acc[g][r];

    __syncthreads();
    if (wave != 0) return;

    // wave 0: reduce 4 partials, gates + cell update.
    // C layout: lane l, reg r -> (batch b = r + 8*(l>=16), hidden col = col0 + (l&15))
    const int ncol = lane & 15;
    const int hcol = col0 + ncol;
    const int boff = (lane >> 4) << 3;
    const float bi = b_ih[0 * HH + hcol] + b_hh[0 * HH + hcol];
    const float bf = b_ih[1 * HH + hcol] + b_hh[1 * HH + hcol];
    const float bg = b_ih[2 * HH + hcol] + b_hh[2 * HH + hcol];
    const float bo = b_ih[3 * HH + hcol] + b_hh[3 * HH + hcol];

#pragma unroll
    for (int r = 0; r < 8; ++r) {
        float gate[4];
#pragma unroll
        for (int g = 0; g < 4; ++g)
            gate[g] = red[0][g][r * 32 + lane] + red[1][g][r * 32 + lane] +
                      red[2][g][r * 32 + lane] + red[3][g][r * 32 + lane];
        int b = r + boff;
        float iv = sigmoidf_(gate[0] + bi);
        float fv = sigmoidf_(gate[1] + bf);
        float gv = tanhf(gate[2] + bg);
        float ov = sigmoidf_(gate[3] + bo);
        size_t idx = (size_t)b * HH + hcol;
        float cn = fv * c[idx] + iv * gv;
        float hn = ov * tanhf(cn);
        c[idx] = cn;
        unsigned short hb = f32_to_bf16(hn);
        h_out[idx] = hb;
        hs_out[idx] = hb;
    }
}

// ---------------- output GEMM: logits[2048,32000] = cat_hs @ W_out.T + b_out ----
// 256 threads = 8 waves in a 2(M) x 4(N) grid; wave tile = 32x64 (acc[2][4]).
// Block tile 64x256. 16 K-steps fully unrolled, double register buffers, and
// sched_group_barrier pins [12 loads (it+1)] -> [8 WMMA (it)] per step.
__global__ __launch_bounds__(256) void out_gemm_kernel(
    const unsigned short* __restrict__ A,    // [2048,512] bf16 (cat_hs)
    const unsigned short* __restrict__ Bw,   // [32000,512] bf16 (W_out)
    const float* __restrict__ bias,          // [32000]
    float* __restrict__ Cout)                // [2048,32000]
{
    const int lane = threadIdx.x & 31;
    const int wave = threadIdx.x >> 5;
    const int wm = wave >> 2;                // 0..1
    const int wn = wave & 3;                 // 0..3
    const int m_base = blockIdx.y * 64 + wm * 32;
    const int n_base = blockIdx.x * 256 + wn * 64;

    v8f acc[2][4];
#pragma unroll
    for (int i = 0; i < 2; ++i)
#pragma unroll
        for (int j = 0; j < 4; ++j) acc[i][j] = v8f{};

    v16bf a_buf[2][2];    // [buffer][m-tile]
    v16bf b_buf[2][4];    // [buffer][n-tile]
#pragma unroll
    for (int i = 0; i < 2; ++i) a_buf[0][i] = load_a_frag(A, HH, m_base + i * 16, 0, lane);
#pragma unroll
    for (int j = 0; j < 4; ++j) b_buf[0][j] = load_b_frag(Bw, HH, n_base + j * 16, 0, lane);
    __builtin_amdgcn_sched_group_barrier(SG_VMEM_READ, 12, 0);   // prologue loads

#pragma unroll
    for (int it = 0; it < 16; ++it) {
        const int cur = it & 1;
        const int nxt = cur ^ 1;
        if (it < 15) {
            const int k0 = (it + 1) * 32;
#pragma unroll
            for (int i = 0; i < 2; ++i)
                a_buf[nxt][i] = load_a_frag(A, HH, m_base + i * 16, k0, lane);
#pragma unroll
            for (int j = 0; j < 4; ++j)
                b_buf[nxt][j] = load_b_frag(Bw, HH, n_base + j * 16, k0, lane);
        }
#pragma unroll
        for (int i = 0; i < 2; ++i)
#pragma unroll
            for (int j = 0; j < 4; ++j)
                acc[i][j] = __builtin_amdgcn_wmma_f32_16x16x32_bf16(
                    false, a_buf[cur][i], false, b_buf[cur][j], (short)0,
                    acc[i][j], false, false);
        if (it < 15) __builtin_amdgcn_sched_group_barrier(SG_VMEM_READ, 12, 0);
        __builtin_amdgcn_sched_group_barrier(SG_WMMA, 8, 0);
    }

    const int ncol = lane & 15;
    const int mtop = m_base + ((lane >> 4) << 3);
#pragma unroll
    for (int j = 0; j < 4; ++j) {
        int n = n_base + j * 16 + ncol;
        float bb = bias[n];
#pragma unroll
        for (int i = 0; i < 2; ++i) {
            int mrow = mtop + i * 16;
#pragma unroll
            for (int r = 0; r < 8; ++r)
                Cout[(size_t)(mrow + r) * VV + n] = acc[i][j][r] + bb;
        }
    }
}

extern "C" void kernel_launch(void* const* d_in, const int* in_sizes, int n_in,
                              void* d_out, int out_size, void* d_ws, size_t ws_size,
                              hipStream_t stream) {
    (void)in_sizes; (void)n_in; (void)out_size; (void)ws_size;
    const int*   xs    = (const int*)d_in[0];
    // d_in[1] = xs_len: unused by the reference computation
    const float* enc_h = (const float*)d_in[2];   // [1,B,H]
    const float* enc_c = (const float*)d_in[3];   // [1,B,H]
    const float* emb   = (const float*)d_in[4];   // [V,E]
    const float* W_ih  = (const float*)d_in[5];   // [4H,E]
    const float* W_hh  = (const float*)d_in[6];   // [4H,H]
    const float* b_ih  = (const float*)d_in[7];   // [4H]
    const float* b_hh  = (const float*)d_in[8];   // [4H]
    const float* W_out = (const float*)d_in[9];   // [V,H]
    const float* b_out = (const float*)d_in[10];  // [V]
    float* logits = (float*)d_out;                // [T*B, V]

    // Workspace layout (bytes)
    char* ws = (char*)d_ws;
    unsigned short* Wih_bf  = (unsigned short*)ws;  ws += (size_t)G4H * EE * 2;      // 2 MB
    unsigned short* Whh_bf  = (unsigned short*)ws;  ws += (size_t)G4H * HH * 2;      // 2 MB
    unsigned short* Wout_bf = (unsigned short*)ws;  ws += (size_t)VV * HH * 2;       // 32 MB
    unsigned short* xemb    = (unsigned short*)ws;  ws += (size_t)TT * BB * EE * 2;  // 2 MB
    unsigned short* cat_hs  = (unsigned short*)ws;  ws += (size_t)TT * BB * HH * 2;  // 2 MB
    unsigned short* hbuf    = (unsigned short*)ws;  ws += (size_t)2 * BB * HH * 2;   // double buffer
    float*          cbuf    = (float*)ws;           ws += (size_t)BB * HH * 4;

    // Convert weights to bf16 (stay L2-resident: 36 MB << 192 MB L2)
    cvt_bf16_kernel<<<2048, 256, 0, stream>>>(W_ih,  Wih_bf,  G4H * EE);
    cvt_bf16_kernel<<<2048, 256, 0, stream>>>(W_hh,  Whh_bf,  G4H * HH);
    cvt_bf16_kernel<<<4096, 256, 0, stream>>>(W_out, Wout_bf, VV * HH);

    // Embedding gather -> time-major bf16 [T,B,E]
    {
        int n = TT * BB * EE;
        embed_kernel<<<(n + 255) / 256, 256, 0, stream>>>(xs, emb, xemb);
    }

    // Init h0 (bf16) and c0 (f32)
    {
        int n = BB * HH;
        init_state_kernel<<<(n + 255) / 256, 256, 0, stream>>>(enc_h, enc_c, hbuf, cbuf, n);
    }

    // Sequential LSTM: one kernel per timestep, h double-buffered across steps.
    for (int t = 0; t < TT; ++t) {
        const unsigned short* h_in  = hbuf + (size_t)(t & 1) * BB * HH;
        unsigned short*       h_out = hbuf + (size_t)((t + 1) & 1) * BB * HH;
        lstm_step_kernel<<<HH / 16, 128, 0, stream>>>(
            xemb + (size_t)t * BB * EE, Wih_bf, Whh_bf, b_ih, b_hh,
            h_in, h_out, cbuf, cat_hs + (size_t)t * BB * HH);
    }

    // Output GEMM: [2048,512] x [512,32000], block tile 64x256
    dim3 grid(VV / 256, (TT * BB) / 64);   // 125 x 32
    out_gemm_kernel<<<grid, 256, 0, stream>>>(cat_hs, Wout_bf, b_out, logits);
}
